// SelfAttention_72877005078973
// MI455X (gfx1250) — matrix operand order repaired and checked by reading, not compiled, not run on hardware
//
#include <hip/hip_runtime.h>
#include <cstdint>

// MI455X / gfx1250, wave32. All matrix math on v_wmma_f32_16x16x32_bf16.

typedef __attribute__((ext_vector_type(16))) __bf16 v16bf;
typedef __attribute__((ext_vector_type(8)))  float  v8f;

#define BB    2
#define NN    1024
#define QDIM  512
#define HH    8
#define DHD   64
#define INNER 512
#define NKEY  2048
#define ASCALE 0.125f   // 1/sqrt(64)

__device__ __forceinline__ unsigned short f2bf(float f) {
  unsigned int u = __float_as_uint(f);
  u += 0x7FFFu + ((u >> 16) & 1u);          // round-to-nearest-even
  return (unsigned short)(u >> 16);
}

union AFrag { v16bf v; unsigned short e[16]; uint4 q[2]; };

// A-operand 16x32 bf16 fragment; `row` points to start of this lane's row (K contiguous).
__device__ __forceinline__ v16bf load_a_frag(const unsigned short* row, int lane, int k0) {
  int koff = (lane >> 4) * 8;
  AFrag f;
  f.q[0] = *(const uint4*)(row + k0 + koff);
  f.q[1] = *(const uint4*)(row + k0 + 16 + koff);
  return f.v;
}
// B-operand 32x16 bf16 fragment; `col` points to start of this lane's column (K contiguous).
__device__ __forceinline__ v16bf load_b_frag(const unsigned short* col, int lane, int k0) {
  int kb = (lane >> 4) * 8;
  AFrag f;
  f.q[0] = *(const uint4*)(col + k0 + kb);
  f.q[1] = *(const uint4*)(col + k0 + 16 + kb);
  return f.v;
}

// ---------------------------------------------------------------- converts
__global__ void k_f2bf(const float* __restrict__ src, unsigned short* __restrict__ dst, int n) {
  int i = blockIdx.x * 256 + threadIdx.x;
  if (i < n) dst[i] = f2bf(src[i]);
}

// nearest-resize 64x64 mask -> 1024 per-token values
__global__ void k_mask(const float* __restrict__ maskIn, float* __restrict__ m) {
  int j = blockIdx.x * 256 + threadIdx.x;
  if (j < NN) {
    int ih = (j >> 5) * 2;
    int iw = (j & 31) * 2;
    m[j] = maskIn[ih * 64 + iw];
  }
}

// ---------------------------------------------------------------- GEMM
// C[M,N] = A[M,K] @ B[N,K]^T ; A,B bf16 row-major (K contiguous), f32 accum.
// epi 0: outF[m*N+n] = c ; epi 1: scatter bf16 into Qbuf (bh,i,d) ; epi 2: outF = c + bias[n]
__global__ void __launch_bounds__(256) k_gemm(const unsigned short* __restrict__ A,
                                              const unsigned short* __restrict__ Bm,
                                              int M, int N, int K, int epi,
                                              float* __restrict__ outF,
                                              unsigned short* __restrict__ outBf,
                                              const float* __restrict__ bias) {
  int lane = threadIdx.x & 31;
  int w    = threadIdx.x >> 5;
  int rowBase = blockIdx.y * 128 + w * 16;
  int colBase = blockIdx.x * 64;
  int ln = lane & 15;

  const unsigned short* arow = A + (size_t)(rowBase + ln) * K;
  const unsigned short* bcol0 = Bm + (size_t)(colBase +  0 + ln) * K;
  const unsigned short* bcol1 = Bm + (size_t)(colBase + 16 + ln) * K;
  const unsigned short* bcol2 = Bm + (size_t)(colBase + 32 + ln) * K;
  const unsigned short* bcol3 = Bm + (size_t)(colBase + 48 + ln) * K;

  v8f accs[4];
  accs[0] = {}; accs[1] = {}; accs[2] = {}; accs[3] = {};

  for (int k0 = 0; k0 < K; k0 += 32) {
    __builtin_prefetch(arow + k0 + 64, 0, 1);     // global_prefetch_b8
    __builtin_prefetch(bcol0 + k0 + 64, 0, 1);
    v16bf a  = load_a_frag(arow, lane, k0);
    v16bf b0 = load_b_frag(bcol0, lane, k0);
    v16bf b1 = load_b_frag(bcol1, lane, k0);
    v16bf b2 = load_b_frag(bcol2, lane, k0);
    v16bf b3 = load_b_frag(bcol3, lane, k0);
    accs[0] = __builtin_amdgcn_wmma_f32_16x16x32_bf16(false, a, false, b0, (short)0, accs[0], false, false);
    accs[1] = __builtin_amdgcn_wmma_f32_16x16x32_bf16(false, a, false, b1, (short)0, accs[1], false, false);
    accs[2] = __builtin_amdgcn_wmma_f32_16x16x32_bf16(false, a, false, b2, (short)0, accs[2], false, false);
    accs[3] = __builtin_amdgcn_wmma_f32_16x16x32_bf16(false, a, false, b3, (short)0, accs[3], false, false);
  }

  int rh = (lane >> 4) * 8;   // C layout: lane>=16 holds rows 8..15
  #pragma unroll
  for (int t = 0; t < 4; ++t) {
    int col = colBase + t * 16 + ln;
    #pragma unroll
    for (int r = 0; r < 8; ++r) {
      int row = rowBase + rh + r;
      float val = accs[t][r];
      if (epi == 0) {
        outF[(size_t)row * N + col] = val;
      } else if (epi == 1) {
        int b = row >> 10, i = row & 1023, h = col >> 6, d = col & 63;
        outBf[((size_t)(b * HH + h) * NN + i) * DHD + d] = f2bf(val);
      } else {
        outF[(size_t)row * N + col] = val + bias[col];
      }
    }
  }
}

// ---------------------------------------------------------------- AdaIN stats
// per channel c: ref mean/std(ddof=1) over 1024 tokens; masked mean/std for feat batch.
__global__ void k_stats(const float* __restrict__ kf, const float* __restrict__ vf,
                        const float* __restrict__ m, float* __restrict__ stats) {
  int c = blockIdx.x;
  int tid = threadIdx.x;
  float a[9] = {0.f,0.f,0.f,0.f,0.f,0.f,0.f,0.f,0.f};
  for (int t = tid; t < NN; t += 256) {
    float mk = m[t];
    float kr = kf[(size_t)t * QDIM + c];
    float vr = vf[(size_t)t * QDIM + c];
    float kx = kf[(size_t)(NN + t) * QDIM + c];
    float vx = vf[(size_t)(NN + t) * QDIM + c];
    a[0] += kr;       a[1] += kr * kr;
    a[2] += vr;       a[3] += vr * vr;
    a[4] += mk * kx;  a[5] += mk * kx * kx;
    a[6] += mk * vx;  a[7] += mk * vx * vx;
    a[8] += mk;
  }
  __shared__ float sh[9][256];
  for (int j = 0; j < 9; ++j) sh[j][tid] = a[j];
  __syncthreads();
  for (int s = 128; s > 0; s >>= 1) {
    if (tid < s)
      for (int j = 0; j < 9; ++j) sh[j][tid] += sh[j][tid + s];
    __syncthreads();
  }
  if (tid == 0) {
    float n = (float)NN;
    float krm = sh[0][0] / n;
    float krs = sqrtf(fmaxf(0.f, (sh[1][0] - n * krm * krm) / (n - 1.f)));
    float vrm = sh[2][0] / n;
    float vrs = sqrtf(fmaxf(0.f, (sh[3][0] - n * vrm * vrm) / (n - 1.f)));
    float cnt = sh[8][0];
    float kfm = sh[4][0] / cnt;
    float kfs = sqrtf(fmaxf(0.f, (sh[5][0] - cnt * kfm * kfm) / (cnt - 1.f)));
    float vfm = sh[6][0] / cnt;
    float vfs = sqrtf(fmaxf(0.f, (sh[7][0] - cnt * vfm * vfm) / (cnt - 1.f)));
    float* o = stats + c * 8;
    o[0] = krm; o[1] = krs; o[2] = kfm; o[3] = kfs;
    o[4] = vrm; o[5] = vrs; o[6] = vfm; o[7] = vfs;
  }
}

// ---------------------------------------------------------------- build K (bh,j,d) and V^T (bh,d,j), bf16
__global__ void k_kvbuild(const float* __restrict__ kf, const float* __restrict__ vf,
                          const float* __restrict__ stats, const float* __restrict__ m,
                          unsigned short* __restrict__ Kb, unsigned short* __restrict__ Vt) {
  int idx = blockIdx.x * 256 + threadIdx.x;        // 16*2048*64
  int bh  = idx >> 17;
  int rem = idx & 131071;
  int j = rem >> 6;
  int d = rem & 63;
  int b = bh >> 3, h = bh & 7;
  int c = h * 64 + d;
  int srcb, t;
  if (j < NN) { srcb = b; t = j; } else { srcb = 0; t = j - NN; }
  size_t si = (size_t)(srcb * NN + t) * QDIM + c;
  float kv = kf[si], vv = vf[si];
  if (b == 1 && j < NN && m[t] > 0.5f) {
    const float* st = stats + c * 8;
    kv = (kv - st[2]) / st[3] * st[1] + st[0];
    vv = (vv - st[6]) / st[7] * st[5] + st[4];
  }
  Kb[(size_t)(bh * NKEY + j) * DHD + d] = f2bf(kv);
  Vt[(size_t)(bh * DHD + d) * NKEY + j] = f2bf(vv);
}

// ---------------------------------------------------------------- fused attention
// 1 wave per (head, 16-query tile). Scores (16x2048 f32) live in LDS; softmax per row;
// P built on the fly as exp(s - rowmax) bf16 A-frags; 1/rowsum folded into epilogue.
__global__ void __launch_bounds__(32) k_attn(const unsigned short* __restrict__ Qb,
                                             const unsigned short* __restrict__ Kb,
                                             const unsigned short* __restrict__ Vt,
                                             const float* __restrict__ m,
                                             unsigned short* __restrict__ attout) {
  extern __shared__ float s[];                 // 16 * 2048 floats = 128 KB
  __shared__ float smax[16], ssum[16];

  int it   = blockIdx.x;        // query tile (0..63)
  int bh   = blockIdx.y;        // head (0..15)
  int lane = threadIdx.x;
  int b = bh >> 3, h = bh & 7;
  int ln   = lane & 15;
  int rh   = (lane >> 4) * 8;   // C-layout row base for this lane half
  int koff = (lane >> 4) * 8;   // A-frag K sub-offset
  // per-row additive mask bias (applies to key cols >= 1024, batch 1 only)
  float bias8[8];
  #pragma unroll
  for (int r = 0; r < 8; ++r) {
    int gi = it * 16 + rh + r;
    bias8[r] = (b == 1 && m[gi] < 0.5f) ? -1.0e9f : 0.0f;
  }

  // Q fragments (A operand, K = 64 -> two frags), kept in registers all kernel
  const unsigned short* qrow = Qb + (size_t)(bh * NN + it * 16 + ln) * DHD;
  v16bf a0 = load_a_frag(qrow, lane, 0);
  v16bf a1 = load_a_frag(qrow, lane, 32);

  // Pass 1: scores -> LDS
  for (int jt = 0; jt < NKEY / 16; ++jt) {
    const unsigned short* krow = Kb + (size_t)(bh * NKEY + jt * 16 + ln) * DHD;
    v16bf b0 = load_b_frag(krow, lane, 0);
    v16bf b1 = load_b_frag(krow, lane, 32);
    v8f c = {};
    c = __builtin_amdgcn_wmma_f32_16x16x32_bf16(false, a0, false, b0, (short)0, c, false, false);
    c = __builtin_amdgcn_wmma_f32_16x16x32_bf16(false, a1, false, b1, (short)0, c, false, false);
    #pragma unroll
    for (int r = 0; r < 8; ++r) {
      float val = c[r] * ASCALE;
      if (jt >= 64) val += bias8[r];
      s[(rh + r) * NKEY + jt * 16 + ln] = val;
    }
  }
  __syncthreads();

  // Pass 2: row max & sum. 2 lanes per row scan 1024 floats each with float4 LDS
  // loads (ds_load_b128), combined across the lane pair with shfl_xor(16).
  {
    int row  = ln;
    const float4* srow = (const float4*)(s + row * NKEY + (lane >> 4) * (NKEY / 2));
    float4 m4 = srow[0];
    float mxa = fmaxf(m4.x, m4.y), mxb = fmaxf(m4.z, m4.w);
    for (int j = 1; j < NKEY / 8; ++j) {
      float4 f = srow[j];
      mxa = fmaxf(mxa, fmaxf(f.x, f.y));
      mxb = fmaxf(mxb, fmaxf(f.z, f.w));
    }
    float mx = fmaxf(mxa, mxb);
    mx = fmaxf(mx, __shfl_xor(mx, 16));
    float sa = 0.f, sb = 0.f;
    for (int j = 0; j < NKEY / 8; ++j) {
      float4 f = srow[j];
      sa += __expf(f.x - mx) + __expf(f.y - mx);
      sb += __expf(f.z - mx) + __expf(f.w - mx);
    }
    float sum = sa + sb;
    sum += __shfl_xor(sum, 16);
    if (lane < 16) { smax[row] = mx; ssum[row] = sum; }
  }
  __syncthreads();

  // Pass 3: O = P @ V, P rows = A-frag rows (ln for both lane halves)
  float mymax = smax[ln];
  const float* prow = s + (size_t)ln * NKEY;
  const unsigned short* vbase = Vt + (size_t)bh * DHD * NKEY;
  v8f o[4];
  o[0] = {}; o[1] = {}; o[2] = {}; o[3] = {};

  for (int j0 = 0; j0 < NKEY; j0 += 32) {
    float4 f0 = *(const float4*)(prow + j0 + koff);
    float4 f1 = *(const float4*)(prow + j0 + koff + 4);
    float4 f2 = *(const float4*)(prow + j0 + 16 + koff);
    float4 f3 = *(const float4*)(prow + j0 + 16 + koff + 4);
    float tv[16] = {f0.x, f0.y, f0.z, f0.w, f1.x, f1.y, f1.z, f1.w,
                    f2.x, f2.y, f2.z, f2.w, f3.x, f3.y, f3.z, f3.w};
    AFrag pf;
    #pragma unroll
    for (int e = 0; e < 16; ++e) pf.e[e] = f2bf(__expf(tv[e] - mymax));

    #pragma unroll
    for (int t = 0; t < 4; ++t) {
      const unsigned short* vcol = vbase + (size_t)(t * 16 + ln) * NKEY;
      v16bf bv = load_b_frag(vcol, lane, j0);
      o[t] = __builtin_amdgcn_wmma_f32_16x16x32_bf16(false, pf.v, false, bv, (short)0, o[t], false, false);
    }
  }

  // Epilogue: divide by rowsum, scatter bf16 into (token, h*64+d) layout
  float inv8[8];
  #pragma unroll
  for (int r = 0; r < 8; ++r) inv8[r] = 1.0f / ssum[rh + r];
  int gbase = b * NN + it * 16;
  #pragma unroll
  for (int t = 0; t < 4; ++t) {
    int col = h * DHD + t * 16 + ln;
    #pragma unroll
    for (int r = 0; r < 8; ++r) {
      attout[(size_t)(gbase + rh + r) * INNER + col] = f2bf(o[t][r] * inv8[r]);
    }
  }
}

// ---------------------------------------------------------------- host
extern "C" void kernel_launch(void* const* d_in, const int* in_sizes, int n_in,
                              void* d_out, int out_size, void* d_ws, size_t ws_size,
                              hipStream_t stream) {
  (void)in_sizes; (void)n_in; (void)out_size; (void)ws_size;
  const float* x  = (const float*)d_in[0];
  const float* mk = (const float*)d_in[1];
  const float* Wq = (const float*)d_in[2];
  const float* Wk = (const float*)d_in[3];
  const float* Wv = (const float*)d_in[4];
  const float* Wo = (const float*)d_in[5];
  const float* bo = (const float*)d_in[6];

  char* p = (char*)d_ws;
  size_t off = 0;
  auto alloc = [&](size_t bytes) -> void* {
    void* r = p + off;
    off += (bytes + 255) & ~(size_t)255;
    return r;
  };
  unsigned short* xb   = (unsigned short*)alloc((size_t)2048 * 512 * 2);
  unsigned short* Wqb  = (unsigned short*)alloc((size_t)512 * 512 * 2);
  unsigned short* Wkb  = (unsigned short*)alloc((size_t)512 * 512 * 2);
  unsigned short* Wvb  = (unsigned short*)alloc((size_t)512 * 512 * 2);
  unsigned short* Wob  = (unsigned short*)alloc((size_t)512 * 512 * 2);
  float*          mtok = (float*)alloc((size_t)1024 * 4);
  float*          kf   = (float*)alloc((size_t)2048 * 512 * 4);
  float*          vf   = (float*)alloc((size_t)2048 * 512 * 4);
  unsigned short* Qb   = (unsigned short*)alloc((size_t)16 * 1024 * 64 * 2);
  unsigned short* Kb   = (unsigned short*)alloc((size_t)16 * 2048 * 64 * 2);
  unsigned short* Vt   = (unsigned short*)alloc((size_t)16 * 64 * 2048 * 2);
  float*          st   = (float*)alloc((size_t)512 * 8 * 4);
  unsigned short* ao   = (unsigned short*)alloc((size_t)2048 * 512 * 2);

  k_f2bf<<<(2048 * 512 + 255) / 256, 256, 0, stream>>>(x, xb, 2048 * 512);
  k_f2bf<<<(512 * 512 + 255) / 256, 256, 0, stream>>>(Wq, Wqb, 512 * 512);
  k_f2bf<<<(512 * 512 + 255) / 256, 256, 0, stream>>>(Wk, Wkb, 512 * 512);
  k_f2bf<<<(512 * 512 + 255) / 256, 256, 0, stream>>>(Wv, Wvb, 512 * 512);
  k_f2bf<<<(512 * 512 + 255) / 256, 256, 0, stream>>>(Wo, Wob, 512 * 512);
  k_mask<<<4, 256, 0, stream>>>(mk, mtok);

  dim3 gg(512 / 64, 2048 / 128);
  k_gemm<<<gg, 256, 0, stream>>>(xb, Wqb, 2048, 512, 512, 1, nullptr, Qb, nullptr);
  k_gemm<<<gg, 256, 0, stream>>>(xb, Wkb, 2048, 512, 512, 0, kf, nullptr, nullptr);
  k_gemm<<<gg, 256, 0, stream>>>(xb, Wvb, 2048, 512, 512, 0, vf, nullptr, nullptr);

  k_stats<<<512, 256, 0, stream>>>(kf, vf, mtok, st);
  k_kvbuild<<<(16 * 2048 * 64) / 256, 256, 0, stream>>>(kf, vf, st, mtok, Kb, Vt);

  k_attn<<<dim3(64, 16), 32, 16 * 2048 * sizeof(float), stream>>>(Qb, Kb, Vt, mtok, ao);

  k_gemm<<<gg, 256, 0, stream>>>(ao, Wob, 2048, 512, 512, 2, (float*)d_out, nullptr, bo);
}
